// UniCLLoss_31215822308143
// MI455X (gfx1250) — compile-verified, matrix operand order checked by reading, not run on hardware
//
#include <hip/hip_runtime.h>

#define NN 8192
#define DD 512
#define LOGIT_SCALE 2.659f

typedef __attribute__((ext_vector_type(16))) __bf16 v16bf;
typedef __attribute__((ext_vector_type(8)))  float  v8f;
typedef __attribute__((ext_vector_type(8)))  unsigned short u16x8;

union Frag16 { u16x8 h[2]; v16bf v; };

__device__ __forceinline__ unsigned short f2bf(float x) {
  unsigned u = __float_as_uint(x);
  u += 0x7FFFu + ((u >> 16) & 1u);   // round-to-nearest-even
  return (unsigned short)(u >> 16);
}

// Issue one per-lane 16B async copy: global -> LDS (tracked by ASYNCcnt).
__device__ __forceinline__ void async_b128(unsigned lds_byte, const unsigned short* gptr) {
  asm volatile("global_load_async_to_lds_b128 %0, %1, off"
               :: "v"(lds_byte), "v"(gptr) : "memory");
}

// One block per row: L2-normalize 512 floats, emit bf16.
__global__ void unicl_normalize_bf16(const float* __restrict__ src,
                                     unsigned short* __restrict__ dst) {
  __shared__ float red[256];
  const int row = blockIdx.x;
  const int t = threadIdx.x;
  const float x0 = src[(size_t)row * DD + t];
  const float x1 = src[(size_t)row * DD + t + 256];
  red[t] = x0 * x0 + x1 * x1;
  __syncthreads();
  for (int off = 128; off > 0; off >>= 1) {
    if (t < off) red[t] += red[t + off];
    __syncthreads();
  }
  const float inv = 1.0f / fmaxf(sqrtf(red[0]), 1e-12f);
  dst[(size_t)row * DD + t]       = f2bf(x0 * inv);
  dst[(size_t)row * DD + t + 256] = f2bf(x1 * inv);
}

// 128x128 tile of S = scale * img @ txt^T via v_wmma_f32_16x16x32_bf16.
// Tiles staged with GLOBAL_LOAD_ASYNC_TO_LDS_B128 through a 4-deep LDS ring,
// fused softmax/target statistics (row & column) with atomics.
__global__ __launch_bounds__(256) void unicl_gemm_stats(
    const unsigned short* __restrict__ imgN, const unsigned short* __restrict__ txtN,
    const int* __restrict__ labels,
    float* __restrict__ gRowE, float* __restrict__ gRowW, float* __restrict__ gRowT,
    float* __restrict__ gColE, float* __restrict__ gColW) {
  __shared__ unsigned short TA[4][128 * 40];   // 128 rows x 32 halves, pad to 40
  __shared__ unsigned short TB[4][128 * 40];
  __shared__ float rowE[128], rowW[128], rowT[128], colE[128], colW[128];
  __shared__ int labR[128], labC[128];

  const int tid = threadIdx.x;
  const int i0 = blockIdx.y * 128;   // image rows   (rows of S)
  const int j0 = blockIdx.x * 128;   // text rows    (cols of S)

  if (tid < 128) {
    rowE[tid] = 0.f; rowW[tid] = 0.f; rowT[tid] = 0.f;
    colE[tid] = 0.f; colW[tid] = 0.f;
    labR[tid] = labels[i0 + tid];
    labC[tid] = labels[j0 + tid];
  }

  // Per-thread copy tasks: two 16B chunks of A and of B per K-step.
  const int p0 = tid, p1 = tid + 256;              // 512 chunk tasks per tile
  const int cr0 = p0 >> 2, cs0 = (p0 & 3) * 8;
  const int cr1 = p1 >> 2, cs1 = (p1 & 3) * 8;
  const unsigned la0 = (unsigned)(size_t)&TA[0][cr0 * 40 + cs0];
  const unsigned la1 = (unsigned)(size_t)&TA[0][cr1 * 40 + cs1];
  const unsigned lb0 = (unsigned)(size_t)&TB[0][cr0 * 40 + cs0];
  const unsigned lb1 = (unsigned)(size_t)&TB[0][cr1 * 40 + cs1];
  const unsigned short* gA0 = imgN + (size_t)(i0 + cr0) * DD + cs0;
  const unsigned short* gA1 = imgN + (size_t)(i0 + cr1) * DD + cs1;
  const unsigned short* gB0 = txtN + (size_t)(j0 + cr0) * DD + cs0;
  const unsigned short* gB1 = txtN + (size_t)(j0 + cr1) * DD + cs1;

  const unsigned bufBytes = 128 * 40 * 2;          // 10240 B per tile buffer

  auto issue_batch = [&](int step) {
    const unsigned bo = (unsigned)(step & 3) * bufBytes;
    const int ko = step * 32;
    async_b128(la0 + bo, gA0 + ko);
    async_b128(la1 + bo, gA1 + ko);
    async_b128(lb0 + bo, gB0 + ko);
    async_b128(lb1 + bo, gB1 + ko);
  };

  const int lane = tid & 31;
  const int wave = tid >> 5;
  const int m0 = (wave >> 1) * 32;   // wave's row offset in tile
  const int n0 = (wave & 1) * 64;    // wave's col offset in tile
  const int lrow = lane & 15;
  const int kb = (lane >> 4) * 8;    // K sub-block per half-wave (A/B layout)

  v8f acc[2][4];
  for (int s = 0; s < 2; ++s)
    for (int b = 0; b < 4; ++b)
      acc[s][b] = (v8f){0.f, 0.f, 0.f, 0.f, 0.f, 0.f, 0.f, 0.f};

  auto compute_step = [&](int ks) __attribute__((always_inline)) {
    const unsigned short* As = TA[ks & 3];
    const unsigned short* Bs = TB[ks & 3];
    Frag16 a[2], b[4];
    for (int s = 0; s < 2; ++s) {
      const int base = (m0 + s * 16 + lrow) * 40 + kb;
      a[s].h[0] = *(const u16x8*)&As[base];
      a[s].h[1] = *(const u16x8*)&As[base + 16];
    }
    for (int bb = 0; bb < 4; ++bb) {
      const int base = (n0 + bb * 16 + lrow) * 40 + kb;
      b[bb].h[0] = *(const u16x8*)&Bs[base];
      b[bb].h[1] = *(const u16x8*)&Bs[base + 16];
    }
    for (int s = 0; s < 2; ++s)
      for (int bb = 0; bb < 4; ++bb)
        acc[s][bb] = __builtin_amdgcn_wmma_f32_16x16x32_bf16(
            false, a[s].v, false, b[bb].v, (short)0, acc[s][bb], false, false);
  };

  // Prologue: batches for steps 0 and 1 in flight (ASYNCcnt = 8).
  issue_batch(0);
  issue_batch(1);

  // Steady state: branch-free body.  In-order async retirement: cnt<=4 means
  // batch `ks` has landed (batch ks+1 may still be in flight).  The buffer
  // written by issue_batch(ks+2) was last read at compute(ks-2): two barriers
  // back for every wave -> safe.
#pragma unroll 1
  for (int ks = 0; ks < 14; ++ks) {
    asm volatile("s_wait_asynccnt 0x4" ::: "memory");
    __syncthreads();
    issue_batch(ks + 2);
    compute_step(ks);
  }
  // Tail: nothing left to issue.
  asm volatile("s_wait_asynccnt 0x4" ::: "memory");
  __syncthreads();
  compute_step(14);
  asm volatile("s_wait_asynccnt 0x0" ::: "memory");
  __syncthreads();
  compute_step(15);

  // Epilogue: per-element exp/target; column stats reduce in registers
  // (column is lane-invariant), row stats reduce via shfl_xor butterflies.
  const int half8 = (lane >> 4) * 8;
  float cE[4] = {0.f, 0.f, 0.f, 0.f};
  float cW[4] = {0.f, 0.f, 0.f, 0.f};
  for (int s = 0; s < 2; ++s) {
    for (int r = 0; r < 8; ++r) {
      const int rl = m0 + s * 16 + half8 + r;   // C/D layout: M = r (+8 for hi half)
      const int gi = i0 + rl;
      const int labi = labR[rl];
      float rE = 0.f, rW = 0.f, rT = 0.f;
      for (int bb = 0; bb < 4; ++bb) {
        const int cl = n0 + bb * 16 + lrow;
        const int gj = j0 + cl;
        const int labj = labC[cl];
        const float sv = LOGIT_SCALE * acc[s][bb][r];
        const float e = __expf(sv);
        const float tt = (gi == gj) ? 1.0f : ((labi == labj && labi != -1) ? 1.0f : 0.0f);
        rE += e; rW += tt * sv; rT += tt;
        cE[bb] += e; cW[bb] += tt * sv;
      }
      for (int off = 1; off < 16; off <<= 1) {   // reduce across 16 lanes sharing a row
        rE += __shfl_xor(rE, off, 32);
        rW += __shfl_xor(rW, off, 32);
        rT += __shfl_xor(rT, off, 32);
      }
      if (lrow == 0) {                            // lanes 0 and 16
        atomicAdd(&rowE[rl], rE);
        atomicAdd(&rowW[rl], rW);
        atomicAdd(&rowT[rl], rT);
      }
    }
  }
  for (int bb = 0; bb < 4; ++bb) {
    const int cl = n0 + bb * 16 + lrow;
    atomicAdd(&colE[cl], cE[bb]);
    atomicAdd(&colW[cl], cW[bb]);
  }
  __syncthreads();
  if (tid < 128) {
    atomicAdd(&gRowE[i0 + tid], rowE[tid]);
    atomicAdd(&gRowW[i0 + tid], rowW[tid]);
    atomicAdd(&gRowT[i0 + tid], rowT[tid]);
    atomicAdd(&gColE[j0 + tid], colE[tid]);
    atomicAdd(&gColW[j0 + tid], colW[tid]);
  }
}

// loss = 0.5 * ( mean_i[lse_i - w_i/t_i] + mean_j[lse_j - w_j/t_j] )
__global__ void unicl_final(const float* __restrict__ gRowE, const float* __restrict__ gRowW,
                            const float* __restrict__ gRowT,
                            const float* __restrict__ gColE, const float* __restrict__ gColW,
                            float* __restrict__ out) {
  __shared__ float red[256];
  const int t = threadIdx.x;
  float s = 0.f;
  for (int i = t; i < NN; i += 256) {
    const float tc = gRowT[i];                 // tcnt symmetric: same for row & col i
    s += __logf(gRowE[i]) - gRowW[i] / tc;
    s += __logf(gColE[i]) - gColW[i] / tc;
  }
  red[t] = s;
  __syncthreads();
  for (int off = 128; off > 0; off >>= 1) {
    if (t < off) red[t] += red[t + off];
    __syncthreads();
  }
  if (t == 0) out[0] = red[0] * (0.5f / (float)NN);
}

extern "C" void kernel_launch(void* const* d_in, const int* in_sizes, int n_in,
                              void* d_out, int out_size, void* d_ws, size_t ws_size,
                              hipStream_t stream) {
  const float* text  = (const float*)d_in[0];
  const float* image = (const float*)d_in[1];
  const int* labels  = (const int*)d_in[2];

  // Workspace: [txt bf16 8MB][img bf16 8MB][5 x 8192 f32 accumulators]
  unsigned short* txtN = (unsigned short*)d_ws;
  unsigned short* imgN = txtN + (size_t)NN * DD;
  float* acc = (float*)(imgN + (size_t)NN * DD);
  float* gRowE = acc;
  float* gRowW = acc + NN;
  float* gRowT = acc + 2 * NN;
  float* gColE = acc + 3 * NN;
  float* gColW = acc + 4 * NN;

  unicl_normalize_bf16<<<dim3(NN), 256, 0, stream>>>(text, txtN);
  unicl_normalize_bf16<<<dim3(NN), 256, 0, stream>>>(image, imgN);
  hipMemsetAsync(acc, 0, (size_t)5 * NN * sizeof(float), stream);
  unicl_gemm_stats<<<dim3(NN / 128, NN / 128), 256, 0, stream>>>(
      imgN, txtN, labels, gRowE, gRowW, gRowT, gColE, gColW);
  unicl_final<<<1, 256, 0, stream>>>(gRowE, gRowW, gRowT, gColE, gColW, (float*)d_out);
}
